// SharedGroupLinearLayer_19636590477499
// MI455X (gfx1250) — compile-verified, matrix operand order checked by reading, not run on hardware
//
#include <hip/hip_runtime.h>
#include <hip/hip_bf16.h>
#include <stdint.h>

// ---------------------------------------------------------------------------
// Shapes (fixed by the reference): N=4096 tokens, din=dout=1024, T=8, KD=16
// ---------------------------------------------------------------------------
#define N_TOK 4096
#define DIN   1024
#define DOUT  1024
#define TPL   8
#define KD    16
#define K8    8192   // folded K dimension: T * DIN

typedef __attribute__((ext_vector_type(16))) __bf16 v16bf;
typedef __attribute__((ext_vector_type(8)))  float  v8f;
typedef __attribute__((ext_vector_type(8)))  __bf16 bf16x8;

union Frag16 { v16bf v; bf16x8 h[2]; };

__device__ __forceinline__ float bf2f(uint32_t u16) {
  union { uint32_t i; float f; } x; x.i = u16 << 16; return x.f;
}
__device__ __forceinline__ uint32_t f2bf(float f) {
  union { float f; uint32_t i; } x; x.f = f;
  uint32_t r = x.i + 0x7FFFu + ((x.i >> 16) & 1u);   // round-to-nearest-even
  return r >> 16;
}
__device__ __forceinline__ uint32_t pack_scale(uint32_t u, float s) {
  float lo = bf2f(u & 0xFFFFu) * s;
  float hi = bf2f(u >> 16) * s;
  return (f2bf(hi) << 16) | f2bf(lo);
}

// CDNA5 async copy: 16B global -> LDS, tracked by ASYNCcnt (ISA 10, opcode 98).
// GV mode: per-lane 64-bit global address, per-lane 32-bit LDS offset in VDST.
__device__ __forceinline__ void async_ld16(uint32_t lds_off, const void* gptr) {
  asm volatile("global_load_async_to_lds_b128 %0, %1, off"
               :: "v"(lds_off), "v"((uint64_t)(uintptr_t)gptr)
               : "memory");
}
__device__ __forceinline__ uint32_t lds_off32(const void* p) {
  // generic LDS pointers carry the allocation-relative offset in addr[31:0]
  return (uint32_t)(uintptr_t)p;
}

// Fragment load from row-major bf16 matrix (leading dim `ld` elems, 16B-aligned
// rows). A-frag: lane -> row (lane%16); B-frag from pre-transposed weights:
// lane -> output column. K chunks: half-wave split {0-7,16-23} / {8-15,24-31}.
__device__ __forceinline__ v16bf load_frag_g(const uint16_t* p, int row, int ld, int k0) {
  int lane = threadIdx.x & 31;
  int r  = lane & 15;
  int kb = (lane >> 4) << 3;
  const uint16_t* q = p + (size_t)(row + r) * ld + k0 + kb;
  Frag16 f;
  f.h[0] = *(const bf16x8*)(q);
  f.h[1] = *(const bf16x8*)(q + 16);
  return f.v;
}

// ---------------------------------------------------------------------------
// 1) x (f32) -> xb (bf16), 8 elems/thread
// ---------------------------------------------------------------------------
__global__ void conv_x_kernel(const float* __restrict__ x, uint16_t* __restrict__ xb, int n8) {
  int i = blockIdx.x * blockDim.x + threadIdx.x;
  if (i >= n8) return;
  const float4* p = (const float4*)x + (size_t)i * 2;
  float4 a = p[0], b = p[1];
  uint4 o;
  o.x = (f2bf(a.y) << 16) | f2bf(a.x);
  o.y = (f2bf(a.w) << 16) | f2bf(a.z);
  o.z = (f2bf(b.y) << 16) | f2bf(b.x);
  o.w = (f2bf(b.w) << 16) | f2bf(b.z);
  ((uint4*)xb)[i] = o;
}

// ---------------------------------------------------------------------------
// 2) w (T,din,dout) f32 -> bf16 transpose with parametrized output layout:
//    out[e*estride + t*tstride + d] = w[t,d,e]
//    (estride=DIN,tstride=DOUT*DIN -> [t][e][d];  estride=K8,tstride=DIN -> [e][t*DIN+d])
// ---------------------------------------------------------------------------
__global__ void conv_wT_kernel(const float* __restrict__ w, uint16_t* __restrict__ wbT,
                               size_t estride, size_t tstride) {
  __shared__ float tile[64][65];
  int t = blockIdx.z, d0 = blockIdx.y * 64, e0 = blockIdx.x * 64;
  int c = threadIdx.x & 63, r0 = threadIdx.x >> 6;
  const float* wp = w + (size_t)t * DIN * DOUT;
  #pragma unroll
  for (int j = 0; j < 16; j++) {
    int r = r0 + j * 4;
    tile[r][c] = wp[(size_t)(d0 + r) * DOUT + e0 + c];
  }
  __syncthreads();
  #pragma unroll
  for (int j = 0; j < 16; j++) {
    int r = r0 + j * 4;
    wbT[(size_t)(e0 + r) * estride + (size_t)t * tstride + d0 + c] = (uint16_t)f2bf(tile[c][r]);
  }
}

// ---------------------------------------------------------------------------
// 3) w_read (din,16) f32 -> wrT (16,din) bf16
// ---------------------------------------------------------------------------
__global__ void conv_wread_kernel(const float* __restrict__ wr, uint16_t* __restrict__ wrT) {
  for (int i = threadIdx.x; i < KD * DIN; i += blockDim.x) {
    int kk = i >> 10, d = i & 1023;
    wrT[i] = (uint16_t)f2bf(wr[d * KD + kk]);
  }
}

// ---------------------------------------------------------------------------
// 4) wcombT[t][k][d] = sum_e w[t,d,e] * w_write[t,e,k]   (bf16 out, f32 accum)
// ---------------------------------------------------------------------------
__global__ void wcomb_kernel(const float* __restrict__ w, const float* __restrict__ ww,
                             uint16_t* __restrict__ wcT) {
  __shared__ float sw[64][128];
  __shared__ float sww[128][KD];
  int t = blockIdx.y, d0 = blockIdx.x * 64;
  int tid = threadIdx.x;
  int dl = tid >> 2, kq = (tid & 3) * 4;
  float acc[4] = {0.f, 0.f, 0.f, 0.f};
  const float* wp  = w  + (size_t)t * DIN * DOUT;
  const float* wwp = ww + (size_t)t * DOUT * KD;
  for (int ec = 0; ec < 8; ec++) {
    int eb = ec * 128;
    for (int i = tid; i < 64 * 128; i += 256) {
      int rr = i >> 7, cc = i & 127;
      sw[rr][cc] = wp[(size_t)(d0 + rr) * DOUT + eb + cc];
    }
    for (int i = tid; i < 128 * KD; i += 256) {
      int rr = i >> 4, cc = i & 15;
      sww[rr][cc] = wwp[(size_t)(eb + rr) * KD + cc];
    }
    __syncthreads();
    #pragma unroll 4
    for (int e = 0; e < 128; e++) {
      float wv = sw[dl][e];
      #pragma unroll
      for (int q = 0; q < 4; q++) acc[q] += wv * sww[e][kq + q];
    }
    __syncthreads();
  }
  #pragma unroll
  for (int q = 0; q < 4; q++)
    wcT[((size_t)t * KD + kq + q) * DIN + d0 + dl] = (uint16_t)f2bf(acc[q]);
}

// ---------------------------------------------------------------------------
// 5) attention: x_read = xb@wrT^T, keys_t = xb@wcombT^T, logits row-dot,
//    softmax over T. One wave per 16-token tile; 9 thin WMMA GEMMs per tile.
// ---------------------------------------------------------------------------
__global__ void __launch_bounds__(256) att_kernel(const uint16_t* __restrict__ xb,
                                                  const uint16_t* __restrict__ wrT,
                                                  const uint16_t* __restrict__ wcT,
                                                  float* __restrict__ att) {
  __shared__ float slog[8][16][TPL];
  int tid = threadIdx.x, lane = tid & 31, w = tid >> 5;
  int m0 = (blockIdx.x * 8 + w) * 16;

  v8f accR = {0, 0, 0, 0, 0, 0, 0, 0};
  for (int ks = 0; ks < DIN / 32; ks++) {
    v16bf a = load_frag_g(xb, m0, DIN, ks * 32);
    v16bf b = load_frag_g(wrT, 0, DIN, ks * 32);
    accR = __builtin_amdgcn_wmma_f32_16x16x32_bf16(false, a, false, b, (short)0, accR, false, false);
  }
  for (int t = 0; t < TPL; t++) {
    v8f accK = {0, 0, 0, 0, 0, 0, 0, 0};
    const uint16_t* bp = wcT + (size_t)t * KD * DIN;
    for (int ks = 0; ks < DIN / 32; ks++) {
      v16bf a = load_frag_g(xb, m0, DIN, ks * 32);
      v16bf b = load_frag_g(bp, 0, DIN, ks * 32);
      accK = __builtin_amdgcn_wmma_f32_16x16x32_bf16(false, a, false, b, (short)0, accK, false, false);
    }
    #pragma unroll
    for (int v = 0; v < 8; v++) {
      float p = accR[v] * accK[v];
      p += __shfl_xor(p, 1, 32);
      p += __shfl_xor(p, 2, 32);
      p += __shfl_xor(p, 4, 32);
      p += __shfl_xor(p, 8, 32);
      if ((lane & 15) == 0) slog[w][v + ((lane >> 4) << 3)][t] = p;
    }
  }
  __syncthreads();
  if (lane < 16) {
    float mx = -1e30f;
    #pragma unroll
    for (int t = 0; t < TPL; t++) mx = fmaxf(mx, slog[w][lane][t]);
    float e[TPL], s = 0.f;
    #pragma unroll
    for (int t = 0; t < TPL; t++) { e[t] = __expf(slog[w][lane][t] - mx); s += e[t]; }
    float inv = 1.f / s;
    #pragma unroll
    for (int t = 0; t < TPL; t++) att[(size_t)(m0 + lane) * TPL + t] = e[t] * inv;
  }
}

// ---------------------------------------------------------------------------
// 6a) xs[n][t*DIN+d] = bf16(att[n,t] * x[n,d])  -- hoists all scaling VALU
//     out of the hot GEMM loop. One thread: 8 d-elems, all 8 templates.
// ---------------------------------------------------------------------------
__global__ void scale_x_kernel(const float* __restrict__ x, const float* __restrict__ att,
                               uint16_t* __restrict__ xs) {
  int gid = blockIdx.x * blockDim.x + threadIdx.x;  // 4096*128 = 524288
  int n  = gid >> 7;
  int dc = (gid & 127) << 3;
  const float* xp = x + (size_t)n * DIN + dc;
  float4 a = *(const float4*)xp;
  float4 b = *(const float4*)(xp + 4);
  #pragma unroll
  for (int t = 0; t < TPL; t++) {
    float s = att[(size_t)n * TPL + t];
    uint4 o;
    o.x = (f2bf(a.y * s) << 16) | f2bf(a.x * s);
    o.y = (f2bf(a.w * s) << 16) | f2bf(a.z * s);
    o.z = (f2bf(b.y * s) << 16) | f2bf(b.x * s);
    o.w = (f2bf(b.w * s) << 16) | f2bf(b.z * s);
    *(uint4*)(xs + ((size_t)n * TPL + t) * DIN + dc) = o;
  }
}

// ---------------------------------------------------------------------------
// 6b) pure K=8192 GEMM with double-buffered ASYNC global->LDS staging:
//     out = xs(4096 x 8192) @ wbT2(1024 x 8192)^T
//     128x128 block tile, 8 waves (2x4), 64x32 per wave, BK=32.
// ---------------------------------------------------------------------------
#define BM  128
#define BE  128
#define BK  32
#define LDT 40   // padded LDS pitch (bf16 elems): conflict-free b128 frag loads

__global__ void __launch_bounds__(256) gemm_async_kernel(const uint16_t* __restrict__ xs,
                                                         const uint16_t* __restrict__ wbT2,
                                                         float* __restrict__ out) {
  __shared__ __align__(16) uint16_t sA[2][BM * LDT];
  __shared__ __align__(16) uint16_t sB[2][BE * LDT];

  int tid = threadIdx.x, lane = tid & 31, w = tid >> 5;
  int wm = w >> 2, we = w & 3;
  int m0 = blockIdx.y * BM, e0 = blockIdx.x * BE;

  v8f acc[4][2];
  #pragma unroll
  for (int i = 0; i < 4; i++)
    #pragma unroll
    for (int j = 0; j < 2; j++) acc[i][j] = (v8f){0, 0, 0, 0, 0, 0, 0, 0};

  int srow = tid >> 2;            // staging row 0..63 (and +64)
  int skc  = (tid & 3) * 8;       // 16B chunk within BK
  int r = lane & 15, kb = (lane >> 4) << 3;

  const uint16_t* gA0 = xs   + (size_t)(m0 + srow) * K8 + skc;
  const uint16_t* gB0 = wbT2 + (size_t)(e0 + srow) * K8 + skc;
  const size_t rstep = (size_t)64 * K8;

  // issue one 32-wide K slab (4 async b128 per thread) into buffer `buf`
  auto issue = [&](int ks, int buf) {
    int k0 = ks * BK;
    uint32_t la = lds_off32(&sA[buf][srow * LDT + skc]);
    uint32_t lb = lds_off32(&sB[buf][srow * LDT + skc]);
    async_ld16(la,                 gA0 + k0);
    async_ld16(la + 64 * LDT * 2,  gA0 + rstep + k0);
    async_ld16(lb,                 gB0 + k0);
    async_ld16(lb + 64 * LDT * 2,  gB0 + rstep + k0);
  };

  issue(0, 0);
  issue(1, 1);
  for (int ks = 0; ks < K8 / BK; ks++) {
    int buf = ks & 1;
    // oldest 4 async ops (this slab) must be done; newer slab may stay in flight
    if (ks < K8 / BK - 1) asm volatile("s_wait_asynccnt 0x4" ::: "memory");
    else                  asm volatile("s_wait_asynccnt 0x0" ::: "memory");
    __syncthreads();
    Frag16 fa[4], fb[2];
    #pragma unroll
    for (int i = 0; i < 4; i++) {
      const uint16_t* p = &sA[buf][(wm * 64 + i * 16 + r) * LDT + kb];
      fa[i].h[0] = *(const bf16x8*)p;
      fa[i].h[1] = *(const bf16x8*)(p + 16);
    }
    #pragma unroll
    for (int j = 0; j < 2; j++) {
      const uint16_t* p = &sB[buf][(we * 32 + j * 16 + r) * LDT + kb];
      fb[j].h[0] = *(const bf16x8*)p;
      fb[j].h[1] = *(const bf16x8*)(p + 16);
    }
    #pragma unroll
    for (int i = 0; i < 4; i++)
      #pragma unroll
      for (int j = 0; j < 2; j++)
        acc[i][j] = __builtin_amdgcn_wmma_f32_16x16x32_bf16(
            false, fa[i].v, false, fb[j].v, (short)0, acc[i][j], false, false);
    __syncthreads();                 // all waves done reading buf
    if (ks + 2 < K8 / BK) issue(ks + 2, buf);
  }
  #pragma unroll
  for (int i = 0; i < 4; i++) {
    int grb = m0 + wm * 64 + i * 16 + ((lane >> 4) << 3);
    #pragma unroll
    for (int j = 0; j < 2; j++) {
      int gc = e0 + we * 32 + j * 16 + (lane & 15);
      #pragma unroll
      for (int v = 0; v < 8; v++)
        out[(size_t)(grb + v) * DOUT + gc] = acc[i][j][v];
    }
  }
}

// ---------------------------------------------------------------------------
// 6c) fallback fused GEMM (small workspace): att applied during LDS staging
// ---------------------------------------------------------------------------
__global__ void __launch_bounds__(256) gemm_fused_kernel(const uint16_t* __restrict__ xb,
                                                         const uint16_t* __restrict__ wbT,
                                                         const float* __restrict__ att,
                                                         float* __restrict__ out) {
  __shared__ __align__(16) uint16_t sA[BM * LDT];
  __shared__ __align__(16) uint16_t sB[BE * LDT];
  __shared__ float sAtt[BM];

  int tid = threadIdx.x, lane = tid & 31, w = tid >> 5;
  int wm = w >> 2, we = w & 3;
  int m0 = blockIdx.y * BM, e0 = blockIdx.x * BE;

  v8f acc[4][2];
  #pragma unroll
  for (int i = 0; i < 4; i++)
    #pragma unroll
    for (int j = 0; j < 2; j++) acc[i][j] = (v8f){0, 0, 0, 0, 0, 0, 0, 0};

  int srow = tid >> 2, skc = (tid & 3) * 8;
  int r = lane & 15, kb = (lane >> 4) << 3;

  for (int t = 0; t < TPL; t++) {
    if (tid < BM) sAtt[tid] = att[(size_t)(m0 + tid) * TPL + t];
    __syncthreads();
    const uint16_t* bbase = wbT + (size_t)t * DOUT * DIN;
    for (int ks = 0; ks < DIN / BK; ks++) {
      int k0 = ks * BK;
      #pragma unroll
      for (int h = 0; h < 2; h++) {
        int rr = srow + h * 64;
        uint4 da = *(const uint4*)(xb + (size_t)(m0 + rr) * DIN + k0 + skc);
        float s = sAtt[rr];
        uint4 oa;
        oa.x = pack_scale(da.x, s); oa.y = pack_scale(da.y, s);
        oa.z = pack_scale(da.z, s); oa.w = pack_scale(da.w, s);
        *(uint4*)&sA[rr * LDT + skc] = oa;
        uint4 db = *(const uint4*)(bbase + (size_t)(e0 + rr) * DIN + k0 + skc);
        *(uint4*)&sB[rr * LDT + skc] = db;
      }
      __syncthreads();
      Frag16 fa[4], fb[2];
      #pragma unroll
      for (int i = 0; i < 4; i++) {
        const uint16_t* p = &sA[(wm * 64 + i * 16 + r) * LDT + kb];
        fa[i].h[0] = *(const bf16x8*)p;
        fa[i].h[1] = *(const bf16x8*)(p + 16);
      }
      #pragma unroll
      for (int j = 0; j < 2; j++) {
        const uint16_t* p = &sB[(we * 32 + j * 16 + r) * LDT + kb];
        fb[j].h[0] = *(const bf16x8*)p;
        fb[j].h[1] = *(const bf16x8*)(p + 16);
      }
      #pragma unroll
      for (int i = 0; i < 4; i++)
        #pragma unroll
        for (int j = 0; j < 2; j++)
          acc[i][j] = __builtin_amdgcn_wmma_f32_16x16x32_bf16(
              false, fa[i].v, false, fb[j].v, (short)0, acc[i][j], false, false);
      __syncthreads();
    }
  }
  #pragma unroll
  for (int i = 0; i < 4; i++) {
    int grb = m0 + wm * 64 + i * 16 + ((lane >> 4) << 3);
    #pragma unroll
    for (int j = 0; j < 2; j++) {
      int gc = e0 + we * 32 + j * 16 + (lane & 15);
      #pragma unroll
      for (int v = 0; v < 8; v++)
        out[(size_t)(grb + v) * DOUT + gc] = acc[i][j][v];
    }
  }
}

// ---------------------------------------------------------------------------
// launch
// ---------------------------------------------------------------------------
extern "C" void kernel_launch(void* const* d_in, const int* in_sizes, int n_in,
                              void* d_out, int out_size, void* d_ws, size_t ws_size,
                              hipStream_t stream) {
  const float* x  = (const float*)d_in[0];   // (512,8,1024) = (4096,1024)
  const float* w  = (const float*)d_in[1];   // (8,1024,1024)
  const float* wr = (const float*)d_in[2];   // (1,1024,16)
  const float* ww = (const float*)d_in[3];   // (8,1024,16)
  float* out = (float*)d_out;                // (4096,1024)

  char* ws = (char*)d_ws;
  uint16_t* xb    = (uint16_t*)(ws);                 //  8,388,608 B
  uint16_t* wbT   = (uint16_t*)(ws + 8388608);       // 16,777,216 B
  uint16_t* wcombT= (uint16_t*)(ws + 25165824);      //    262,144 B
  uint16_t* wrT   = (uint16_t*)(ws + 25427968);      //     32,768 B
  float*    attp  = (float*)   (ws + 25460736);      //    131,072 B
  uint16_t* xs    = (uint16_t*)(ws + 25591808);      // 67,108,864 B (big path)
  const size_t need_big = 25591808ull + 67108864ull; // 92,700,672 B

  bool big = (ws_size >= need_big);

  conv_x_kernel    <<<(N_TOK * DIN / 8 + 255) / 256, 256, 0, stream>>>(x, xb, N_TOK * DIN / 8);
  if (big)  // layout [e][t*DIN+d] for the K=8192 GEMM
    conv_wT_kernel <<<dim3(16, 16, 8), 256, 0, stream>>>(w, wbT, (size_t)K8, (size_t)DIN);
  else      // layout [t][e][d] for the fused GEMM
    conv_wT_kernel <<<dim3(16, 16, 8), 256, 0, stream>>>(w, wbT, (size_t)DIN, (size_t)DOUT * DIN);
  conv_wread_kernel<<<1, 256, 0, stream>>>(wr, wrT);
  wcomb_kernel     <<<dim3(16, 8), 256, 0, stream>>>(w, ww, wcombT);
  att_kernel       <<<N_TOK / (16 * 8), 256, 0, stream>>>(xb, wrT, wcombT, attp);

  if (big) {
    scale_x_kernel   <<<(N_TOK * DIN / 8 + 255) / 256, 256, 0, stream>>>(x, attp, xs);
    gemm_async_kernel<<<dim3(DOUT / BE, N_TOK / BM), 256, 0, stream>>>(xs, wbT, out);
  } else {
    gemm_fused_kernel<<<dim3(DOUT / BE, N_TOK / BM), 256, 0, stream>>>(xb, wbT, attp, out);
  }
}
